// Graph_Attention_62199716381004
// MI455X (gfx1250) — matrix operand order repaired and checked by reading, not compiled, not run on hardware
//
#include <hip/hip_runtime.h>
#include <hip/hip_bf16.h>

#define SLOPE 0.2f
#define KNN 20

typedef _Float16 v16h_t __attribute__((ext_vector_type(16)));
typedef _Float16 h2_t   __attribute__((ext_vector_type(2)));
typedef float    v8f_t  __attribute__((ext_vector_type(8)));

__device__ __forceinline__ float leaky(float v) { return v > 0.f ? v : SLOPE * v; }

// ---------------------------------------------------------------------------
// Transpose x (B,3,N) -> xT (B,N,3)
// ---------------------------------------------------------------------------
__global__ void transpose_in(const float* __restrict__ x, float* __restrict__ xT, int N) {
    int t = blockIdx.x * blockDim.x + threadIdx.x;   // over B*N
    int b = t / N, n = t % N;
#pragma unroll
    for (int c = 0; c < 3; ++c)
        xT[(size_t)t * 3 + c] = x[((size_t)b * 3 + c) * N + n];
}

// ---------------------------------------------------------------------------
// KNN: one thread per query point, query features in registers,
// neighbor tiles staged through LDS (float4 wide loads), next-tile prefetch
// (global_prefetch_b8), fully-unrolled top-20 sorted insert kept in VGPRs.
// ---------------------------------------------------------------------------
template <int C>
__global__ void knn_kernel(const float* __restrict__ xT, int* __restrict__ idx, int N) {
    constexpr int TILE = 128;
    __shared__ float tile[TILE * C];

    int blocksPerBatch = N / 256;
    int b  = blockIdx.x / blocksPerBatch;
    int qn = (blockIdx.x % blocksPerBatch) * 256 + threadIdx.x;
    const float* xb = xT + (size_t)b * N * C;

    float qf[C];
#pragma unroll
    for (int c = 0; c < C; ++c) qf[c] = xb[(size_t)qn * C + c];

    float bestd[KNN];
    int   besti[KNN];
#pragma unroll
    for (int k = 0; k < KNN; ++k) { bestd[k] = 1e30f; besti[k] = 0; }

    for (int t0 = 0; t0 < N; t0 += TILE) {
        __syncthreads();
        if constexpr (C % 4 == 0) {
            const float4* src = (const float4*)(xb + (size_t)t0 * C);
            float4*       dst = (float4*)tile;
            for (int i = threadIdx.x; i < TILE * C / 4; i += blockDim.x)
                dst[i] = src[i];
        } else {
            for (int i = threadIdx.x; i < TILE * C; i += blockDim.x)
                tile[i] = xb[(size_t)t0 * C + i];
        }
        // Prefetch next candidate tile into cache while we crunch this one
        if (t0 + TILE < N)
            __builtin_prefetch(xb + (size_t)(t0 + TILE) * C + (size_t)threadIdx.x * 16, 0, 0);
        __syncthreads();
        for (int j = 0; j < TILE; ++j) {
            float d = 0.f;
#pragma unroll
            for (int c = 0; c < C; ++c) {
                float df = qf[c] - tile[j * C + c];
                d = fmaf(df, df, d);
            }
            if (d < bestd[KNN - 1]) {
                bestd[KNN - 1] = d; besti[KNN - 1] = t0 + j;
#pragma unroll
                for (int p = KNN - 1; p > 0; --p) {
                    if (bestd[p] < bestd[p - 1]) {
                        float td = bestd[p]; bestd[p] = bestd[p - 1]; bestd[p - 1] = td;
                        int   ti = besti[p]; besti[p] = besti[p - 1]; besti[p - 1] = ti;
                    }
                }
            }
        }
    }
#pragma unroll
    for (int k = 0; k < KNN; ++k)
        idx[((size_t)b * N + qn) * KNN + k] = besti[k];
}

// ---------------------------------------------------------------------------
// WMMA fragment loaders (16-bit A 16x32, B 32x16; layouts per CDNA5 ISA 7.12.2)
// base pre-offset to tile origin; ldk = padded K stride (halves).
// ---------------------------------------------------------------------------
__device__ __forceinline__ v16h_t load_frag_a(const _Float16* base, int ldk, int lane) {
    v16h_t f;
    int m    = lane & 15;
    int koff = (lane & 16) ? 8 : 0;   // upper half-wave holds K+8 (and K+24)
#pragma unroll
    for (int r = 0; r < 8; ++r) {
        int k = ((r < 4) ? (2 * r) : (16 + 2 * (r - 4))) + koff;
        h2_t v = *(const h2_t*)(base + m * ldk + k);
        f[2 * r] = v[0]; f[2 * r + 1] = v[1];
    }
    return f;
}

__device__ __forceinline__ v16h_t load_frag_b(const _Float16* base, int ldk, int lane) {
    v16h_t f;
    int n    = lane & 15;
    int koff = (lane & 16) ? 8 : 0;   // upper half-wave holds K+8 (and K+24)
#pragma unroll
    for (int r = 0; r < 8; ++r) {
        int k = ((r < 4) ? (2 * r) : (16 + 2 * (r - 4))) + koff;
        h2_t v = *(const h2_t*)(base + n * ldk + k);
        f[2 * r] = v[0]; f[2 * r + 1] = v[1];
    }
    return f;
}

// ---------------------------------------------------------------------------
// Fused GAT layer: per wave = one query point.
//  - gather 20 neighbor rows (prefetch k+1 while converting k)
//  - build 32 x 2C edge-feature tile (f16) in LDS (rows 20..31 zero-padded)
//  - hw = A @ W via v_wmma_f32_16x16x32_f16, staged to LDS
//  - leaky -> per-head logits -> softmax over k -> weighted sum -> leaky
// ---------------------------------------------------------------------------
template <int CIN, int COUT>
__global__ void gat_kernel(const float* __restrict__ xT, const int* __restrict__ idx,
                           const float* __restrict__ W, const float* __restrict__ a,
                           float* __restrict__ out, int N) {
    constexpr int C2    = 2 * CIN;
    constexpr int KP    = (C2 + 31) & ~31;   // padded K (32 or 128)
    constexpr int H     = COUT / 4;
    constexpr int WAVES = 4;

    extern __shared__ char smem[];
    _Float16* sW   = (_Float16*)smem;                       // [COUT][KP]  (W transposed)
    _Float16* sA   = sW + COUT * KP;                        // per wave [32][KP]
    float*    sHW  = (float*)(sA + WAVES * 32 * KP);        // per wave [32][COUT]
    float*    sCtr = sHW + WAVES * 32 * COUT;               // per wave [CIN]

    int wave = threadIdx.x >> 5;
    int lane = threadIdx.x & 31;

    // Cooperative load of W^T into LDS as f16, zero-padded K
    for (int i = threadIdx.x; i < COUT * KP; i += blockDim.x) {
        int n = i / KP, k = i % KP;
        sW[i] = (k < C2) ? (_Float16)W[(size_t)k * COUT + n] : (_Float16)0.f;
    }

    int q = blockIdx.x * WAVES + wave;       // global query in [0, B*N)
    int b = q / N, n = q % N;
    const float* xb = xT + (size_t)b * N * CIN;

    _Float16* myA   = sA + wave * 32 * KP;
    float*    myHW  = sHW + wave * 32 * COUT;
    float*    myCtr = sCtr + wave * CIN;

    for (int c = lane; c < CIN; c += 32) myCtr[c] = xb[(size_t)n * CIN + c];
    for (int i = lane; i < 32 * KP; i += 32) myA[i] = (_Float16)0.f;
    __syncthreads();

    const int* myIdx = idx + (size_t)q * KNN;
    int jnext = myIdx[0];
    __builtin_prefetch(xb + (size_t)jnext * CIN, 0, 0);
    for (int k = 0; k < KNN; ++k) {
        int j = jnext;
        if (k + 1 < KNN) {
            jnext = myIdx[k + 1];
            __builtin_prefetch(xb + (size_t)jnext * CIN, 0, 0);  // hide gather latency
        }
        const float* nb = xb + (size_t)j * CIN;
        for (int c = lane; c < C2; c += 32) {
            float v = (c < CIN) ? (nb[c] - myCtr[c]) : myCtr[c - CIN];
            myA[k * KP + c] = (_Float16)v;
        }
    }
    __syncthreads();

    // GEMM: hw(32 x COUT) = A(32 x KP) * W(KP x COUT); EXEC full, uniform loops
#pragma unroll
    for (int nt = 0; nt < COUT / 16; ++nt) {
#pragma unroll
        for (int mt = 0; mt < 2; ++mt) {
            v8f_t acc = {};
#pragma unroll
            for (int kt = 0; kt < KP / 32; ++kt) {
                v16h_t af = load_frag_a(myA + mt * 16 * KP + kt * 32, KP, lane);
                v16h_t bf = load_frag_b(sW + nt * 16 * KP + kt * 32, KP, lane);
                acc = __builtin_amdgcn_wmma_f32_16x16x32_f16(
                    false, af, false, bf, (short)0, acc, false, false);
            }
            // D layout: lane holds column nt*16 + lane%16; VGPR r = row r (+8 upper)
            int ncol  = nt * 16 + (lane & 15);
            int rbase = mt * 16 + ((lane & 16) ? 8 : 0);
#pragma unroll
            for (int r = 0; r < 8; ++r)
                myHW[(rbase + r) * COUT + ncol] = acc[r];
        }
    }
    __syncthreads();

    // Attention: one lane per head (H <= 32)
    if (lane < H) {
        int h = lane;
        float av[4];
#pragma unroll
        for (int d = 0; d < 4; ++d) av[d] = a[d * H + h];

        float logits[KNN];
        float mx = -1e30f;
#pragma unroll
        for (int k = 0; k < KNN; ++k) {
            float s = 0.f;
#pragma unroll
            for (int d = 0; d < 4; ++d)
                s = fmaf(leaky(myHW[k * COUT + h * 4 + d]), av[d], s);
            logits[k] = s;
            mx = fmaxf(mx, s);
        }
        float denom = 0.f;
#pragma unroll
        for (int k = 0; k < KNN; ++k) { logits[k] = __expf(logits[k] - mx); denom += logits[k]; }
        float inv = 1.f / denom;

        float o[4] = {0.f, 0.f, 0.f, 0.f};
#pragma unroll
        for (int k = 0; k < KNN; ++k) {
            float w = logits[k] * inv;
#pragma unroll
            for (int d = 0; d < 4; ++d)
                o[d] = fmaf(w, myHW[k * COUT + h * 4 + d], o[d]);
        }
#pragma unroll
        for (int d = 0; d < 4; ++d)
            out[(size_t)q * COUT + h * 4 + d] = leaky(o[d]);
    }
}

// ---------------------------------------------------------------------------
// BatchNorm stats: one block per channel, reduce sum / sumsq over B*N
// ---------------------------------------------------------------------------
__global__ void bn_stats(const float* __restrict__ in, float* __restrict__ mean,
                         float* __restrict__ rstd, int NP, int COUT) {
    int c = blockIdx.x;
    __shared__ float ss[256], ss2[256];
    float s = 0.f, s2 = 0.f;
    for (int q = threadIdx.x; q < NP; q += blockDim.x) {
        float v = in[(size_t)q * COUT + c];
        s += v; s2 = fmaf(v, v, s2);
    }
    ss[threadIdx.x] = s; ss2[threadIdx.x] = s2;
    __syncthreads();
    for (int st = 128; st > 0; st >>= 1) {
        if (threadIdx.x < st) {
            ss[threadIdx.x]  += ss[threadIdx.x + st];
            ss2[threadIdx.x] += ss2[threadIdx.x + st];
        }
        __syncthreads();
    }
    if (threadIdx.x == 0) {
        float mu  = ss[0] / (float)NP;
        float var = ss2[0] / (float)NP - mu * mu;
        mean[c] = mu;
        rstd[c] = rsqrtf(var + 1e-5f);
    }
}

// ---------------------------------------------------------------------------
// BN apply + leaky; writes xT for next layer and/or final (B,COUT,N) output
// ---------------------------------------------------------------------------
__global__ void bn_apply(const float* __restrict__ in, const float* __restrict__ mean,
                         const float* __restrict__ rstd, const float* __restrict__ g,
                         const float* __restrict__ bt, float* __restrict__ xTnext,
                         float* __restrict__ fout, int NP, int COUT, int N) {
    int t = blockIdx.x * blockDim.x + threadIdx.x;
    if (t >= NP * COUT) return;
    int q = t / COUT, c = t % COUT;
    float v = (in[t] - mean[c]) * rstd[c] * g[c] + bt[c];
    v = leaky(v);
    if (xTnext) xTnext[t] = v;
    if (fout) {
        int b = q / N, n = q % N;
        fout[((size_t)b * COUT + c) * N + n] = v;
    }
}

// ---------------------------------------------------------------------------
// Host launcher
// ---------------------------------------------------------------------------
static inline size_t gat_smem(int KP, int CIN, int COUT) {
    const int WAVES = 4;
    return (size_t)COUT * KP * 2 + (size_t)WAVES * 32 * KP * 2 +
           (size_t)WAVES * 32 * COUT * 4 + (size_t)WAVES * CIN * 4;
}

extern "C" void kernel_launch(void* const* d_in, const int* in_sizes, int n_in,
                              void* d_out, int out_size, void* d_ws, size_t ws_size,
                              hipStream_t stream) {
    constexpr int B = 8, N = 2048, BN = B * N;
    const float* x  = (const float*)d_in[0];
    const float* W0 = (const float*)d_in[1];
    const float* a0 = (const float*)d_in[2];
    const float* g0 = (const float*)d_in[3];
    const float* b0 = (const float*)d_in[4];
    const float* W1 = (const float*)d_in[5];
    const float* a1 = (const float*)d_in[6];
    const float* g1 = (const float*)d_in[7];
    const float* b1 = (const float*)d_in[8];
    const float* W2 = (const float*)d_in[9];
    const float* a2 = (const float*)d_in[10];
    const float* g2 = (const float*)d_in[11];
    const float* b2 = (const float*)d_in[12];

    float* ws    = (float*)d_ws;
    float* xTa   = ws;                                   // BN * 128
    float* xTb   = xTa + (size_t)BN * 128;               // BN * 128
    int*   idxb  = (int*)(xTb + (size_t)BN * 128);       // BN * KNN
    float* outb  = (float*)(idxb + (size_t)BN * KNN);    // BN * 128
    float* meanb = outb + (size_t)BN * 128;              // 128
    float* rstdb = meanb + 128;                          // 128

    transpose_in<<<BN / 256, 256, 0, stream>>>(x, xTa, N);

    // ---- Layer 0: 3 -> 64 ----
    knn_kernel<3><<<BN / 256, 256, 0, stream>>>(xTa, idxb, N);
    gat_kernel<3, 64><<<BN / 4, 128, gat_smem(32, 3, 64), stream>>>(xTa, idxb, W0, a0, outb, N);
    bn_stats<<<64, 256, 0, stream>>>(outb, meanb, rstdb, BN, 64);
    bn_apply<<<(BN * 64 + 255) / 256, 256, 0, stream>>>(outb, meanb, rstdb, g0, b0,
                                                        xTb, nullptr, BN, 64, N);

    // ---- Layer 1: 64 -> 64 ----
    knn_kernel<64><<<BN / 256, 256, 0, stream>>>(xTb, idxb, N);
    gat_kernel<64, 64><<<BN / 4, 128, gat_smem(128, 64, 64), stream>>>(xTb, idxb, W1, a1, outb, N);
    bn_stats<<<64, 256, 0, stream>>>(outb, meanb, rstdb, BN, 64);
    bn_apply<<<(BN * 64 + 255) / 256, 256, 0, stream>>>(outb, meanb, rstdb, g1, b1,
                                                        xTa, nullptr, BN, 64, N);

    // ---- Layer 2: 64 -> 128, final output (B, 128, N) ----
    knn_kernel<64><<<BN / 256, 256, 0, stream>>>(xTa, idxb, N);
    gat_kernel<64, 128><<<BN / 4, 128, gat_smem(128, 64, 128), stream>>>(xTa, idxb, W2, a2, outb, N);
    bn_stats<<<128, 256, 0, stream>>>(outb, meanb, rstdb, BN, 128);
    bn_apply<<<(BN * 128 + 255) / 256, 256, 0, stream>>>(outb, meanb, rstdb, g2, b2,
                                                         nullptr, (float*)d_out, BN, 128, N);
}
